// TransformerBlock_23562190586200
// MI455X (gfx1250) — compile-verified, matrix operand order checked by reading, not run on hardware
//
#include <hip/hip_runtime.h>
#include <math.h>

// ---------------------------------------------------------------------------
// MI455X (gfx1250) transformer block.
// Compute-bound: ~258 GFLOP of GEMMs vs ~0.5 GB HBM traffic (~20us @ 23.3TB/s).
// All matmuls on v_wmma_f32_16x16x32_bf16 (bf16 operands, f32 accumulate).
// 64MB score matrix is materialized (fits the 192MB L2) -> no flash attention.
// GEMM: block tile 256x128, BK=32, 8 waves, wave tile 64x64 = 16 WMMAs/step.
// Data movement: GLOBAL_LOAD_ASYNC_TO_LDS_B128 double buffering (zero VGPR
// staging, ASYNCcnt-tracked) when the toolchain exposes the builtins;
// otherwise single-buffer staging through transient VGPRs.
// ---------------------------------------------------------------------------

typedef __attribute__((ext_vector_type(8)))  float  v8f;
typedef __attribute__((ext_vector_type(8)))  __bf16 v8bf;
typedef __attribute__((ext_vector_type(16))) __bf16 v16bf;

#if defined(__gfx1250__) && \
    __has_builtin(__builtin_amdgcn_global_load_async_to_lds_b128) && \
    __has_builtin(__builtin_amdgcn_s_wait_asynccnt)
#define USE_ASYNC 1
#else
#define USE_ASYNC 0
#endif

// builtin parameter types: int4-vector pointers in global (AS1) / LDS (AS3)
typedef int v4i_raw __attribute__((__vector_size__(4 * sizeof(int))));
typedef __attribute__((address_space(1))) v4i_raw glb_v4i_t;
typedef __attribute__((address_space(3))) v4i_raw lds_v4i_t;

__device__ __forceinline__ glb_v4i_t* glb_cast(const void* p) {
  return (glb_v4i_t*)(unsigned long long)p;   // launder const via inttoptr
}
__device__ __forceinline__ lds_v4i_t* lds_cast(void* p) {
  // flat address of an LDS object: low 32 bits are the LDS byte offset
  return (lds_v4i_t*)(unsigned)(unsigned long long)p;
}

__device__ __forceinline__ __bf16 to_bf16(float f) {
  unsigned u = __builtin_bit_cast(unsigned, f);
  u += 0x7fffu + ((u >> 16) & 1u);               // round-to-nearest-even
  unsigned short h = (unsigned short)(u >> 16);
  return __builtin_bit_cast(__bf16, h);
}

#define LDSK 40   // 32 + 8 pad (80B row stride: staggers banks, keeps 16B align)
#define ASZ  (256 * LDSK)
#define BSZ  (128 * LDSK)

// C[M,N] = A[M,K] * B_T[N,K]  (B stored N-major, K-contiguous)
// epi: 0=QKV split, 1=f32 store, 2=+residual f32, 3=+bias+GELU bf16, 4=+bias+res f32
__global__ __launch_bounds__(256) void gemm_bf16_kernel(
    const __bf16* __restrict__ A, const __bf16* __restrict__ B,
    long lda, long ldb, int K, int N,
    long sA, long sB, long sO, long sR,
    void* out0v, void* out1v, void* out2v,
    const float* __restrict__ bias, const float* __restrict__ residIn,
    int epi)
{
#if USE_ASYNC
  __shared__ __align__(16) __bf16 lA[2 * ASZ];
  __shared__ __align__(16) __bf16 lB[2 * BSZ];
#else
  __shared__ __align__(16) __bf16 lA[ASZ];
  __shared__ __align__(16) __bf16 lB[BSZ];
#endif

  const int tid  = threadIdx.x;
  const int wave = tid >> 5;
  const int ln   = tid & 31;
  const int wm   = wave & 3;     // 4 waves along M (64 rows each)
  const int wn   = wave >> 2;    // 2 waves along N (64 cols each)

  const long z = blockIdx.z;
  A += z * sA;
  B += z * sB;
  const float* resid = residIn ? (residIn + z * sR) : (const float*)0;

  const long m0 = (long)blockIdx.y * 256;
  const long n0 = (long)blockIdx.x * 128;

  v8f acc[4][4] = {};

  // loader mapping: 16B chunks; A tile 256x32 -> 4 chunks/thread, B 128x32 -> 2
  const int lrow = tid >> 2;          // 0..63
  const int co   = (tid & 3) * 8;     // 0,8,16,24
  const __bf16* pa = A + (m0 + lrow) * lda + co;
  const __bf16* pb = B + (n0 + lrow) * ldb + co;
  const long stepA = 64 * lda;
  const long stepB = 64 * ldb;
  const int ldsOff = lrow * LDSK + co;

  // fragment lane striping (ISA 16-bit A/B layout)
  const int frow = ln & 15;
  const int ko   = (ln < 16) ? 0 : 8;

#if USE_ASYNC
  // ---- async DMA double-buffered pipeline: zero VGPR staging ----
  #pragma unroll
  for (int i = 0; i < 4; ++i)
    __builtin_amdgcn_global_load_async_to_lds_b128(
        glb_cast(pa + i * stepA), lds_cast(&lA[ldsOff + i * 64 * LDSK]), 0, 0);
  #pragma unroll
  for (int i = 0; i < 2; ++i)
    __builtin_amdgcn_global_load_async_to_lds_b128(
        glb_cast(pb + i * stepB), lds_cast(&lB[ldsOff + i * 64 * LDSK]), 0, 0);

  int cur = 0;
  for (int k0 = 0; k0 < K; k0 += 32) {
    __builtin_amdgcn_s_wait_asynccnt(0);   // my slab-k0 portion landed in LDS
    __syncthreads();                       // everyone's landed; prev reads done
    const bool more = (k0 + 32) < K;
    if (more) {                            // DMA next slab into the other buffer
      const int nb = cur ^ 1;
      #pragma unroll
      for (int i = 0; i < 4; ++i)
        __builtin_amdgcn_global_load_async_to_lds_b128(
            glb_cast(pa + k0 + 32 + i * stepA),
            lds_cast(&lA[nb * ASZ + ldsOff + i * 64 * LDSK]), 0, 0);
      #pragma unroll
      for (int i = 0; i < 2; ++i)
        __builtin_amdgcn_global_load_async_to_lds_b128(
            glb_cast(pb + k0 + 32 + i * stepB),
            lds_cast(&lB[nb * BSZ + ldsOff + i * 64 * LDSK]), 0, 0);
      __builtin_prefetch(pa + k0 + 64, 0, 1);   // global_prefetch_b8: L2 warm
      __builtin_prefetch(pb + k0 + 64, 0, 1);
    }

    const __bf16* bufA = &lA[cur * ASZ];
    const __bf16* bufB = &lB[cur * BSZ];

    v16bf bfv[4];
    #pragma unroll
    for (int nt = 0; nt < 4; ++nt) {
      const __bf16* p = &bufB[(wn * 64 + nt * 16 + frow) * LDSK + ko];
      v8bf lo = *(const v8bf*)p;
      v8bf hi = *(const v8bf*)(p + 16);
      bfv[nt] = __builtin_shufflevector(lo, hi, 0,1,2,3,4,5,6,7,8,9,10,11,12,13,14,15);
    }
    #pragma unroll
    for (int mt = 0; mt < 4; ++mt) {
      const __bf16* p = &bufA[(wm * 64 + mt * 16 + frow) * LDSK + ko];
      v8bf lo = *(const v8bf*)p;
      v8bf hi = *(const v8bf*)(p + 16);
      v16bf af = __builtin_shufflevector(lo, hi, 0,1,2,3,4,5,6,7,8,9,10,11,12,13,14,15);
      #pragma unroll
      for (int nt = 0; nt < 4; ++nt)
        acc[mt][nt] = __builtin_amdgcn_wmma_f32_16x16x32_bf16(
            false, af, false, bfv[nt], (short)0, acc[mt][nt], false, false);
    }
    cur ^= 1;
  }
#else
  // ---- fallback: single buffer, transient VGPR staging, two barriers ----
  for (int k0 = 0; k0 < K; k0 += 32) {
    uint4 ra[4], rb[2];
    #pragma unroll
    for (int i = 0; i < 4; ++i) ra[i] = *(const uint4*)(pa + k0 + i * stepA);
    #pragma unroll
    for (int i = 0; i < 2; ++i) rb[i] = *(const uint4*)(pb + k0 + i * stepB);
    #pragma unroll
    for (int i = 0; i < 4; ++i) *(uint4*)(&lA[ldsOff + i * 64 * LDSK]) = ra[i];
    #pragma unroll
    for (int i = 0; i < 2; ++i) *(uint4*)(&lB[ldsOff + i * 64 * LDSK]) = rb[i];
    __syncthreads();

    v16bf bfv[4];
    #pragma unroll
    for (int nt = 0; nt < 4; ++nt) {
      const __bf16* p = &lB[(wn * 64 + nt * 16 + frow) * LDSK + ko];
      v8bf lo = *(const v8bf*)p;
      v8bf hi = *(const v8bf*)(p + 16);
      bfv[nt] = __builtin_shufflevector(lo, hi, 0,1,2,3,4,5,6,7,8,9,10,11,12,13,14,15);
    }
    #pragma unroll
    for (int mt = 0; mt < 4; ++mt) {
      const __bf16* p = &lA[(wm * 64 + mt * 16 + frow) * LDSK + ko];
      v8bf lo = *(const v8bf*)p;
      v8bf hi = *(const v8bf*)(p + 16);
      v16bf af = __builtin_shufflevector(lo, hi, 0,1,2,3,4,5,6,7,8,9,10,11,12,13,14,15);
      #pragma unroll
      for (int nt = 0; nt < 4; ++nt)
        acc[mt][nt] = __builtin_amdgcn_wmma_f32_16x16x32_bf16(
            false, af, false, bfv[nt], (short)0, acc[mt][nt], false, false);
    }
    __syncthreads();
  }
#endif

  // ---- epilogue ----
  const int ln15 = ln & 15;
  const int mhi  = (ln >> 4) * 8;
  float*  outF = (float*)out0v;
  if (epi == 1 || epi == 2) outF += z * sO;
  __bf16* outH = (__bf16*)out0v;

  #pragma unroll
  for (int mt = 0; mt < 4; ++mt) {
    #pragma unroll
    for (int nt = 0; nt < 4; ++nt) {
      const long mbase = m0 + wm * 64 + mt * 16 + mhi;
      const long n     = n0 + wn * 64 + nt * 16 + ln15;
      #pragma unroll
      for (int rr = 0; rr < 8; ++rr) {
        const long m = mbase + rr;
        float v = acc[mt][nt][rr];
        if (epi == 0) {                       // QKV split; V stored transposed [B][D][S]
          __bf16* q  = (__bf16*)out0v;
          __bf16* kk = (__bf16*)out1v;
          __bf16* vt = (__bf16*)out2v;
          if (n < 1024)       q[m * 1024 + n] = to_bf16(v);
          else if (n < 2048)  kk[m * 1024 + (n - 1024)] = to_bf16(v);
          else {
            const long bb = m >> 11, s = m & 2047;
            vt[((bb << 10) + (n - 2048)) * 2048 + s] = to_bf16(v);
          }
        } else if (epi == 1) {                // raw f32 scores
          outF[m * (long)N + n] = v;
        } else if (epi == 2) {                // attn + residual
          outF[m * (long)N + n] = v + resid[m * (long)N + n];
        } else if (epi == 3) {                // fc1 + bias + exact GELU -> bf16
          float x = v + bias[n];
          float g = 0.5f * x * (1.0f + erff(x * 0.70710678118654752440f));
          outH[m * (long)N + n] = to_bf16(g);
        } else {                              // fc2 + bias + residual -> f32 out
          outF[m * (long)N + n] = v + bias[n] + resid[m * (long)N + n];
        }
      }
    }
  }
}

// ---------------------------------------------------------------------------
// LayerNorm over D=1024, one 256-thread block per row, f32 in -> bf16 out
// ---------------------------------------------------------------------------
__global__ __launch_bounds__(256) void layernorm_kernel(
    const float* __restrict__ x, const float* __restrict__ g,
    const float* __restrict__ b, __bf16* __restrict__ out)
{
  __shared__ float red[256];
  const long row = blockIdx.x;
  const int tid = threadIdx.x;
  const int i = tid * 4;

  const float4 v = *(const float4*)(x + row * 1024 + i);
  float s = v.x + v.y + v.z + v.w;
  red[tid] = s; __syncthreads();
  for (int o = 128; o > 0; o >>= 1) { if (tid < o) red[tid] += red[tid + o]; __syncthreads(); }
  const float mean = red[0] * (1.0f / 1024.0f);
  __syncthreads();

  const float dx = v.x - mean, dy = v.y - mean, dz = v.z - mean, dw = v.w - mean;
  red[tid] = dx*dx + dy*dy + dz*dz + dw*dw; __syncthreads();
  for (int o = 128; o > 0; o >>= 1) { if (tid < o) red[tid] += red[tid + o]; __syncthreads(); }
  const float rstd = rsqrtf(red[0] * (1.0f / 1024.0f) + 1e-5f);

  const float4 gg = *(const float4*)(g + i);
  const float4 bb = *(const float4*)(b + i);
  __bf16* o4 = out + row * 1024 + i;
  o4[0] = to_bf16(dx * rstd * gg.x + bb.x);
  o4[1] = to_bf16(dy * rstd * gg.y + bb.y);
  o4[2] = to_bf16(dz * rstd * gg.z + bb.z);
  o4[3] = to_bf16(dw * rstd * gg.w + bb.w);
}

// ---------------------------------------------------------------------------
// Row softmax over 2048 cols, f32 scores -> bf16 probs, one block per row
// ---------------------------------------------------------------------------
__global__ __launch_bounds__(256) void softmax_kernel(
    const float* __restrict__ sc, __bf16* __restrict__ p)
{
  __shared__ float red[256];
  const long row = blockIdx.x;
  const int tid = threadIdx.x;

  float v[8];
  float mx = -3.0e38f;
  #pragma unroll
  for (int i = 0; i < 8; ++i) {
    v[i] = sc[row * 2048 + tid + i * 256];
    mx = fmaxf(mx, v[i]);
  }
  red[tid] = mx; __syncthreads();
  for (int o = 128; o > 0; o >>= 1) { if (tid < o) red[tid] = fmaxf(red[tid], red[tid + o]); __syncthreads(); }
  mx = red[0]; __syncthreads();

  float sum = 0.0f;
  #pragma unroll
  for (int i = 0; i < 8; ++i) { v[i] = __expf(v[i] - mx); sum += v[i]; }
  red[tid] = sum; __syncthreads();
  for (int o = 128; o > 0; o >>= 1) { if (tid < o) red[tid] += red[tid + o]; __syncthreads(); }
  const float inv = 1.0f / red[0];

  #pragma unroll
  for (int i = 0; i < 8; ++i)
    p[row * 2048 + tid + i * 256] = to_bf16(v[i] * inv);
}

// ---------------------------------------------------------------------------
// Cast f32 [R][C] -> bf16 [C][R]  (weights repacked to N-major once per call)
// ---------------------------------------------------------------------------
__global__ __launch_bounds__(256) void cast_transpose_kernel(
    const float* __restrict__ in, __bf16* __restrict__ out, int R, int C)
{
  const long idx = (long)blockIdx.x * 256 + threadIdx.x;
  const long rr = idx / C, cc = idx % C;
  out[cc * (long)R + rr] = to_bf16(in[idx]);
}

// ---------------------------------------------------------------------------
extern "C" void kernel_launch(void* const* d_in, const int* in_sizes, int n_in,
                              void* d_out, int out_size, void* d_ws, size_t ws_size,
                              hipStream_t stream) {
  (void)in_sizes; (void)n_in; (void)out_size; (void)ws_size;
  const float* hidden = (const float*)d_in[0];
  const float* w_qkv  = (const float*)d_in[1];
  const float* fc1_w  = (const float*)d_in[2];
  const float* fc1_b  = (const float*)d_in[3];
  const float* fc2_w  = (const float*)d_in[4];
  const float* fc2_b  = (const float*)d_in[5];
  const float* ln1_g  = (const float*)d_in[6];
  const float* ln1_b  = (const float*)d_in[7];
  const float* ln2_g  = (const float*)d_in[8];
  const float* ln2_b  = (const float*)d_in[9];
  float* out = (float*)d_out;

  const long S = 2048, D = 1024, F = 4096, BS = 4 * S;

  char* w = (char*)d_ws;
  auto alloc = [&](size_t bytes) -> char* {
    char* p = w; w += (bytes + 255) & ~(size_t)255; return p;
  };
  __bf16* wqkvT  = (__bf16*)alloc((size_t)3 * D * D * 2);    // [3D][D]
  __bf16* fc1T   = (__bf16*)alloc((size_t)F * D * 2);        // [F][D]
  __bf16* fc2T   = (__bf16*)alloc((size_t)D * F * 2);        // [D][F]
  __bf16* hln1   = (__bf16*)alloc((size_t)BS * D * 2);
  __bf16* Q      = (__bf16*)alloc((size_t)BS * D * 2);
  __bf16* Km     = (__bf16*)alloc((size_t)BS * D * 2);
  __bf16* Vt     = (__bf16*)alloc((size_t)BS * D * 2);       // [B][D][S]
  float*  scores = (float*) alloc((size_t)4 * S * S * 4);    // 64MB (fits L2)
  __bf16* probs  = (__bf16*)alloc((size_t)4 * S * S * 2);
  float*  h2     = (float*) alloc((size_t)BS * D * 4);
  __bf16* hln2   = (__bf16*)alloc((size_t)BS * D * 2);
  __bf16* act    = (__bf16*)alloc((size_t)BS * F * 2);

  // weight repack (f32 -> bf16, transposed to [N][K])
  cast_transpose_kernel<<<(1024L*3072)/256, 256, 0, stream>>>(w_qkv, wqkvT, 1024, 3072);
  cast_transpose_kernel<<<(1024L*4096)/256, 256, 0, stream>>>(fc1_w, fc1T, 1024, 4096);
  cast_transpose_kernel<<<(4096L*1024)/256, 256, 0, stream>>>(fc2_w, fc2T, 4096, 1024);

  // LN1
  layernorm_kernel<<<BS, 256, 0, stream>>>(hidden, ln1_g, ln1_b, hln1);

  // QKV : [8192,1024] x [1024,3072]
  gemm_bf16_kernel<<<dim3(3072/128, BS/256, 1), 256, 0, stream>>>(
      hln1, wqkvT, D, D, (int)D, 3072, 0, 0, 0, 0,
      (void*)Q, (void*)Km, (void*)Vt, nullptr, nullptr, 0);

  // scores = Q K^T per batch : [2048,2048], K = 1024
  gemm_bf16_kernel<<<dim3(S/128, S/256, 4), 256, 0, stream>>>(
      Q, Km, D, D, (int)D, (int)S, S * D, S * D, S * S, 0,
      (void*)scores, nullptr, nullptr, nullptr, nullptr, 1);

  // softmax rows
  softmax_kernel<<<BS, 256, 0, stream>>>(scores, probs);

  // attn = P V + residual : [2048,1024], K = 2048
  gemm_bf16_kernel<<<dim3(D/128, S/256, 4), 256, 0, stream>>>(
      probs, Vt, S, S, (int)S, (int)D, S * S, D * S, S * D, S * D,
      (void*)h2, nullptr, nullptr, nullptr, hidden, 2);

  // LN2
  layernorm_kernel<<<BS, 256, 0, stream>>>(h2, ln2_g, ln2_b, hln2);

  // FC1 + bias + GELU : [8192,4096], K = 1024
  gemm_bf16_kernel<<<dim3(F/128, BS/256, 1), 256, 0, stream>>>(
      hln2, fc1T, D, D, (int)D, (int)F, 0, 0, 0, 0,
      (void*)act, nullptr, nullptr, fc1_b, nullptr, 3);

  // FC2 + bias + residual -> d_out : [8192,1024], K = 4096
  gemm_bf16_kernel<<<dim3(D/128, BS/256, 1), 256, 0, stream>>>(
      act, fc2T, F, F, (int)F, (int)D, 0, 0, 0, 0,
      (void*)out, nullptr, nullptr, fc2_b, h2, 4);
}
